// GraphDecoder_14319420965000
// MI455X (gfx1250) — matrix operand order repaired and checked by reading, not compiled, not run on hardware
//
#include <hip/hip_runtime.h>
#include <hip/hip_bf16.h>

typedef __attribute__((ext_vector_type(16))) _Float16 v16h;
typedef __attribute__((ext_vector_type(8)))  _Float16 v8h;
typedef __attribute__((ext_vector_type(4)))  _Float16 v4h;
typedef __attribute__((ext_vector_type(8)))  float    v8f;
typedef __attribute__((ext_vector_type(4)))  float    v4f;

#define TILE_M 64
#define TILE_N 64
#define KC     32
#define LDAS   40   // LDS stride (halves) for A tiles: 80B rows -> conflict-free b128 frag loads
#define LDBS   40   // LDS stride (halves) for B tiles (column-major)

__device__ __forceinline__ v4h cvt4(v4f x) {
    v4h h;
    h[0] = (_Float16)x[0]; h[1] = (_Float16)x[1];
    h[2] = (_Float16)x[2]; h[3] = (_Float16)x[3];
    return h;
}

// D = act(A[M,K] @ W[K,N] + bias + residual). f16 WMMA, f32 accumulate.
// Shapes compile-time; M via grid.x. Pipelined: chunk k+1 global loads overlap chunk k WMMAs.
template<int N, int K, bool BIAS, bool RES, bool RELU>
__global__ __launch_bounds__(256) void wmma_gemm(
    const float* __restrict__ A, const float* __restrict__ W,
    const float* __restrict__ bias, const float* __restrict__ residual,
    float* __restrict__ C)
{
    constexpr int NC = K / KC;
    __shared__ _Float16 lA[TILE_M * LDAS];
    __shared__ _Float16 lB[TILE_N * LDBS];

    const int tid  = threadIdx.x;
    const int wave = tid >> 5;
    const int lane = tid & 31;
    const int row0 = blockIdx.x * TILE_M;
    const int col0 = blockIdx.y * TILE_N;
    const int tr   = (wave & 3) * 16;        // wave's row tile inside block
    const int tc   = (wave >> 2) * 32;       // wave's two col tiles: tc, tc+16
    const int nf   = lane & 15;              // fragment row (A) / col (B,C)
    const int kbA  = (lane >> 4) << 3;       // A-frag K base: 0 or 8
    const int kbB  = (lane >> 4) << 4;       // B-frag K base: 0 or 16

    // staging coords: 512 float4 groups per tile, 2 per thread
    const int gA0 = tid, gA1 = tid + 256;
    const int ar0 = gA0 >> 3, ac0 = (gA0 & 7) << 2;   // A: row, col4
    const int ar1 = gA1 >> 3, ac1 = (gA1 & 7) << 2;
    const int bk0 = gA0 >> 4, bn0 = (gA0 & 15) << 2;  // B: k, n4
    const int bk1 = gA1 >> 4, bn1 = (gA1 & 15) << 2;

    v4f ra0, ra1, rb0, rb1;

    // prime chunk 0
    ra0 = *(const v4f*)&A[(row0 + ar0) * K + ac0];
    ra1 = *(const v4f*)&A[(row0 + ar1) * K + ac1];
    rb0 = *(const v4f*)&W[bk0 * N + col0 + bn0];
    rb1 = *(const v4f*)&W[bk1 * N + col0 + bn1];

    v8f acc0 = {}; v8f acc1 = {};

#pragma unroll
    for (int kc = 0; kc < NC; ++kc) {
        // stage regs -> LDS (f16)
        *(v4h*)&lA[ar0 * LDAS + ac0] = cvt4(ra0);
        *(v4h*)&lA[ar1 * LDAS + ac1] = cvt4(ra1);
        v4h hb0 = cvt4(rb0), hb1 = cvt4(rb1);
#pragma unroll
        for (int i = 0; i < 4; ++i) {
            lB[(bn0 + i) * LDBS + bk0] = hb0[i];
            lB[(bn1 + i) * LDBS + bk1] = hb1[i];
        }
        __syncthreads();

        // issue next chunk's global loads; they fly during the WMMAs below
        if (kc + 1 < NC) {
            const int k0 = (kc + 1) * KC;
            ra0 = *(const v4f*)&A[(row0 + ar0) * K + k0 + ac0];
            ra1 = *(const v4f*)&A[(row0 + ar1) * K + k0 + ac1];
            rb0 = *(const v4f*)&W[(k0 + bk0) * N + col0 + bn0];
            rb1 = *(const v4f*)&W[(k0 + bk1) * N + col0 + bn1];
        }

        // A fragment per ISA layout: e<8 -> K=kbA+e ; e>=8 -> K=16+kbA+(e-8)
        const _Float16* pa = &lA[(tr + nf) * LDAS + kbA];
        v8h a_lo = *(const v8h*)(pa);
        v8h a_hi = *(const v8h*)(pa + 16);
        v16h afrag = __builtin_shufflevector(a_lo, a_hi,
            0,1,2,3,4,5,6,7,8,9,10,11,12,13,14,15);
        // B fragment: same K order as A: e<8 -> K=kbA+e ; e>=8 -> K=16+kbA+(e-8)
        const _Float16* pb0 = &lB[(tc + nf) * LDBS + kbA];
        const _Float16* pb1 = &lB[(tc + 16 + nf) * LDBS + kbA];
        v16h bf0 = __builtin_shufflevector(*(const v8h*)(pb0), *(const v8h*)(pb0 + 16),
            0,1,2,3,4,5,6,7,8,9,10,11,12,13,14,15);
        v16h bf1 = __builtin_shufflevector(*(const v8h*)(pb1), *(const v8h*)(pb1 + 16),
            0,1,2,3,4,5,6,7,8,9,10,11,12,13,14,15);

        acc0 = __builtin_amdgcn_wmma_f32_16x16x32_f16(false, afrag, false, bf0,
                                                      (short)0, acc0, false, false);
        acc1 = __builtin_amdgcn_wmma_f32_16x16x32_f16(false, afrag, false, bf1,
                                                      (short)0, acc1, false, false);
        __syncthreads();
    }

    const float b0 = BIAS ? bias[col0 + tc + nf]      : 0.f;
    const float b1 = BIAS ? bias[col0 + tc + 16 + nf] : 0.f;
    const int mbase = row0 + tr + ((lane >> 4) << 3);
    const int c0 = col0 + tc + nf;
    const int c1 = c0 + 16;
#pragma unroll
    for (int r = 0; r < 8; ++r) {
        int m = mbase + r;
        float x0 = acc0[r] + b0;
        float x1 = acc1[r] + b1;
        if (RES) { x0 += residual[m * N + c0]; x1 += residual[m * N + c1]; }
        if (RELU) { x0 = fmaxf(x0, 0.f); x1 = fmaxf(x1, 0.f); }
        C[m * N + c0] = x0;
        C[m * N + c1] = x1;
    }
}

// One block per (b, head). 64 tokens, dh=32. scores->softmax->attn@v, probs to d_out.
__global__ __launch_bounds__(256) void attn_kernel(
    const float* __restrict__ q, const float* __restrict__ k,
    const float* __restrict__ v, float* __restrict__ attO,
    float* __restrict__ attn_out)
{
    const int b = blockIdx.x >> 2, head = blockIdx.x & 3;
    __shared__ _Float16 lQ[64 * LDAS];
    __shared__ _Float16 lK[64 * LDAS];
    __shared__ float    sS[64 * 68];
    __shared__ float    sRed[256];
    __shared__ float    sSum[256];
    __shared__ _Float16 lP[64 * 72];
    __shared__ _Float16 lV[32 * 72];

    const int tid  = threadIdx.x;
    const int wave = tid >> 5;
    const int lane = tid & 31;
    const int nf   = lane & 15;
    const int kbA  = (lane >> 4) << 3;
    const int kbB  = (lane >> 4) << 4;
    const int hb   = head * 32;

    // stage q,k row-major f16 (v4f loads); v column-major f16
#pragma unroll
    for (int it = 0; it < 2; ++it) {
        int g = tid + it * 256;              // 512 groups: t = g>>3, d4 = (g&7)*4
        int t = g >> 3, d4 = (g & 7) << 2;
        v4f qv = *(const v4f*)&q[(b * 64 + t) * 128 + hb + d4];
        v4f kv = *(const v4f*)&k[(b * 64 + t) * 128 + hb + d4];
        v4f vv = *(const v4f*)&v[(b * 64 + t) * 128 + hb + d4];
        *(v4h*)&lQ[t * LDAS + d4] = cvt4(qv);
        *(v4h*)&lK[t * LDAS + d4] = cvt4(kv);
        v4h hv = cvt4(vv);
#pragma unroll
        for (int i = 0; i < 4; ++i) lV[(d4 + i) * 72 + t] = hv[i];
    }
    __syncthreads();

    // scores = (q @ k^T) / sqrt(32): 4x4 tiles of 16x16, wave -> 2 tiles
    {
        const int tr = (wave & 3) * 16;
        const int tc = (wave >> 2) * 32;
        v8f s0 = {}; v8f s1 = {};
        const _Float16* pa = &lQ[(tr + nf) * LDAS + kbA];
        v8h a_lo = *(const v8h*)(pa);
        v8h a_hi = *(const v8h*)(pa + 16);
        v16h afrag = __builtin_shufflevector(a_lo, a_hi,
            0,1,2,3,4,5,6,7,8,9,10,11,12,13,14,15);
        const _Float16* pb0 = &lK[(tc + nf) * LDAS + kbA];
        const _Float16* pb1 = &lK[(tc + 16 + nf) * LDAS + kbA];
        v16h bf0 = __builtin_shufflevector(*(const v8h*)(pb0), *(const v8h*)(pb0 + 16),
            0,1,2,3,4,5,6,7,8,9,10,11,12,13,14,15);
        v16h bf1 = __builtin_shufflevector(*(const v8h*)(pb1), *(const v8h*)(pb1 + 16),
            0,1,2,3,4,5,6,7,8,9,10,11,12,13,14,15);
        s0 = __builtin_amdgcn_wmma_f32_16x16x32_f16(false, afrag, false, bf0,
                                                    (short)0, s0, false, false);
        s1 = __builtin_amdgcn_wmma_f32_16x16x32_f16(false, afrag, false, bf1,
                                                    (short)0, s1, false, false);
        const float scale = 0.17677669529663689f; // 1/sqrt(32)
        const int mrow = tr + ((lane >> 4) << 3);
#pragma unroll
        for (int r = 0; r < 8; ++r) {
            sS[(mrow + r) * 68 + tc + nf]      = s0[r] * scale;
            sS[(mrow + r) * 68 + tc + 16 + nf] = s1[r] * scale;
        }
    }
    __syncthreads();

    // parallel softmax: 4 lanes per row, 16 elements each
    const int srow = tid >> 2;            // 0..63
    const int sq   = tid & 3;             // quarter
    float* srp = &sS[srow * 68 + sq * 16];
    {
        float mx = -3.4e38f;
#pragma unroll
        for (int j = 0; j < 16; ++j) mx = fmaxf(mx, srp[j]);
        sRed[tid] = mx;
    }
    __syncthreads();
    float mx4;
    {
        const float* r4 = &sRed[srow * 4];
        mx4 = fmaxf(fmaxf(r4[0], r4[1]), fmaxf(r4[2], r4[3]));
        float sum = 0.f;
#pragma unroll
        for (int j = 0; j < 16; ++j) {
            float e = __expf(srp[j] - mx4);
            srp[j] = e; sum += e;
        }
        sSum[tid] = sum;
    }
    __syncthreads();
    {
        const float* s4 = &sSum[srow * 4];
        float inv = 1.f / (s4[0] + s4[1] + s4[2] + s4[3]);
        size_t base = (size_t)(b * 4 + head) * 4096 + srow * 64 + sq * 16;
        _Float16* lpp = &lP[srow * 72 + sq * 16];
#pragma unroll
        for (int j = 0; j < 16; ++j) {
            float p = srp[j] * inv;
            attn_out[base + j] = p;
            lpp[j] = (_Float16)p;
        }
    }
    __syncthreads();

    // out = attn[64x64] @ v[64x32]: 8 tiles, one per wave, K=64 in 2 chunks
    {
        const int tm = (wave & 3) * 16;
        const int tn = (wave >> 2) * 16;
        v8f o = {};
#pragma unroll
        for (int k0 = 0; k0 < 64; k0 += 32) {
            const _Float16* pa = &lP[(tm + nf) * 72 + k0 + kbA];
            v8h a_lo = *(const v8h*)(pa);
            v8h a_hi = *(const v8h*)(pa + 16);
            v16h afrag = __builtin_shufflevector(a_lo, a_hi,
                0,1,2,3,4,5,6,7,8,9,10,11,12,13,14,15);
            const _Float16* pb = &lV[(tn + nf) * 72 + k0 + kbA];
            v16h bfrag = __builtin_shufflevector(*(const v8h*)(pb), *(const v8h*)(pb + 16),
                0,1,2,3,4,5,6,7,8,9,10,11,12,13,14,15);
            o = __builtin_amdgcn_wmma_f32_16x16x32_f16(false, afrag, false, bfrag,
                                                       (short)0, o, false, false);
        }
        const int mrow = tm + ((lane >> 4) << 3);
#pragma unroll
        for (int r = 0; r < 8; ++r)
            attO[(b * 64 + mrow + r) * 128 + hb + tn + nf] = o[r];
    }
}

// edge_logits[b,i,j] (i<j): we2 . relu(P_i + Q_j + be1) + be2, mirrored; diag = -1e9
__global__ __launch_bounds__(256) void edge_kernel(
    const float* __restrict__ P, const float* __restrict__ Q,
    const float* __restrict__ be1, const float* __restrict__ we2,
    const float* __restrict__ be2, float* __restrict__ edge)
{
    const int b = blockIdx.x;
    __shared__ float sP[64 * 129];
    __shared__ float sQ[32 * 129];
    __shared__ float sW[128];
    __shared__ float sB[128];
    const int tid = threadIdx.x;

#pragma unroll
    for (int it = 0; it < 8; ++it) {           // 2048 float4 groups
        int g = tid + it * 256;
        int r = g >> 5, c4 = (g & 31) << 2;
        v4f x = *(const v4f*)&P[b * 8192 + r * 128 + c4];
#pragma unroll
        for (int i = 0; i < 4; ++i) sP[r * 129 + c4 + i] = x[i];
    }
    if (tid < 128) { sW[tid] = we2[tid]; sB[tid] = be1[tid]; }
    const float b2 = be2[0];

    for (int jh = 0; jh < 64; jh += 32) {
        __syncthreads();
#pragma unroll
        for (int it = 0; it < 4; ++it) {       // 1024 float4 groups
            int g = tid + it * 256;
            int r = g >> 5, c4 = (g & 31) << 2;
            v4f x = *(const v4f*)&Q[b * 8192 + (jh + r) * 128 + c4];
#pragma unroll
            for (int i = 0; i < 4; ++i) sQ[r * 129 + c4 + i] = x[i];
        }
        __syncthreads();
#pragma unroll
        for (int it = 0; it < 8; ++it) {
            int idx = tid + it * 256;
            int i = idx >> 5, j = jh + (idx & 31);
            if (i < j) {
                const float* pi = &sP[i * 129];
                const float* pj = &sQ[(idx & 31) * 129];
                float s = 0.f;
#pragma unroll 16
                for (int kk = 0; kk < 128; ++kk)
                    s += sW[kk] * fmaxf(pi[kk] + pj[kk] + sB[kk], 0.f);
                float w = s + b2;
                edge[b * 4096 + i * 64 + j] = w;
                edge[b * 4096 + j * 64 + i] = w;
            } else if (i == j) {
                edge[b * 4096 + i * 65] = -1e9f;
            }
        }
    }
}

// node_features = h[8192,128] @ wf[128,7] + bf
__global__ __launch_bounds__(256) void nf_kernel(
    const float* __restrict__ h, const float* __restrict__ wf,
    const float* __restrict__ bf, float* __restrict__ nf)
{
    __shared__ float sH[32 * 129];
    __shared__ float sWF[128 * 7];
    const int tid = threadIdx.x;
    const int row0 = blockIdx.x * 32;
#pragma unroll
    for (int it = 0; it < 4; ++it) {           // 1024 float4 groups
        int g = tid + it * 256;
        int r = g >> 5, c4 = (g & 31) << 2;
        v4f x = *(const v4f*)&h[(row0 + r) * 128 + c4];
#pragma unroll
        for (int i = 0; i < 4; ++i) sH[r * 129 + c4 + i] = x[i];
    }
#pragma unroll
    for (int it = 0; it < 4; ++it) {
        int e = tid + it * 256;
        if (e < 896) sWF[e] = wf[e];
    }
    __syncthreads();
    if (tid < 224) {
        int r = tid / 7, f = tid % 7;
        float s = 0.f;
#pragma unroll 16
        for (int kk = 0; kk < 128; ++kk) s += sH[r * 129 + kk] * sWF[kk * 7 + f];
        nf[(row0 + r) * 7 + f] = s + bf[f];
    }
}

template<int N, int K, bool BIAS, bool RES, bool RELU>
static inline void launch_gemm(const float* A, const float* W, const float* bias,
                               const float* residual, float* C, int M, hipStream_t s)
{
    dim3 grid(M / TILE_M, N / TILE_N);
    wmma_gemm<N, K, BIAS, RES, RELU><<<grid, dim3(256), 0, s>>>(A, W, bias, residual, C);
}

extern "C" void kernel_launch(void* const* d_in, const int* in_sizes, int n_in,
                              void* d_out, int out_size, void* d_ws, size_t ws_size,
                              hipStream_t stream)
{
    const float* z   = (const float*)d_in[0];
    const float* w1  = (const float*)d_in[1];  const float* b1  = (const float*)d_in[2];
    const float* w2  = (const float*)d_in[3];  const float* b2  = (const float*)d_in[4];
    const float* wn  = (const float*)d_in[5];  const float* bn  = (const float*)d_in[6];
    const float* wq1 = (const float*)d_in[7];  const float* bq1 = (const float*)d_in[8];
    const float* wk1 = (const float*)d_in[9];  const float* bk1 = (const float*)d_in[10];
    const float* wv1 = (const float*)d_in[11]; const float* bv1 = (const float*)d_in[12];
    const float* wo1 = (const float*)d_in[13]; const float* bo1 = (const float*)d_in[14];
    const float* wq2 = (const float*)d_in[15]; const float* bq2 = (const float*)d_in[16];
    const float* wk2 = (const float*)d_in[17]; const float* bk2 = (const float*)d_in[18];
    const float* wv2 = (const float*)d_in[19]; const float* bv2 = (const float*)d_in[20];
    const float* wo2 = (const float*)d_in[21]; const float* bo2 = (const float*)d_in[22];
    const float* wc1 = (const float*)d_in[23]; const float* bc1 = (const float*)d_in[24];
    const float* wc2 = (const float*)d_in[25]; const float* bc2 = (const float*)d_in[26];
    const float* wf  = (const float*)d_in[27]; const float* bf  = (const float*)d_in[28];
    const float* we1 = (const float*)d_in[29]; const float* be1 = (const float*)d_in[30];
    const float* we2 = (const float*)d_in[31]; const float* be2 = (const float*)d_in[32];

    float* out = (float*)d_out;
    float* nf    = out;                 // [128,64,7]
    float* edge  = out + 57344;         // [128,64,64]
    float* attn1 = out + 581632;        // [128,4,64,64]
    float* attn2 = out + 2678784;       // [128,4,64,64]

    float* ws   = (float*)d_ws;
    float* h    = ws;                   // [8192,128]
    float* qb   = ws + 1 * 1048576;
    float* kb   = ws + 2 * 1048576;
    float* vb   = ws + 3 * 1048576;
    float* aO   = ws + 4 * 1048576;
    float* tlat = ws + 5 * 1048576;     // [128,256]
    float* g    = tlat + 32768;         // [128,128]

    // latent MLP -> g
    launch_gemm<256, 128, true, false, true >(z, w1, b1, nullptr, tlat, 128, stream);
    launch_gemm<128, 256, true, false, true >(tlat, w2, b2, nullptr, g, 128, stream);
    // node projection: row-major [128,8192] == reshaped [8192,128]
    launch_gemm<8192, 128, true, false, false>(g, wn, bn, nullptr, h, 128, stream);

    // MHA layer 1
    launch_gemm<128, 128, true, false, false>(h, wq1, bq1, nullptr, qb, 8192, stream);
    launch_gemm<128, 128, true, false, false>(h, wk1, bk1, nullptr, kb, 8192, stream);
    launch_gemm<128, 128, true, false, false>(h, wv1, bv1, nullptr, vb, 8192, stream);
    attn_kernel<<<dim3(512), dim3(256), 0, stream>>>(qb, kb, vb, aO, attn1);
    launch_gemm<128, 128, true, true,  false>(aO, wo1, bo1, h, h, 8192, stream);

    // MHA layer 2
    launch_gemm<128, 128, true, false, false>(h, wq2, bq2, nullptr, qb, 8192, stream);
    launch_gemm<128, 128, true, false, false>(h, wk2, bk2, nullptr, kb, 8192, stream);
    launch_gemm<128, 128, true, false, false>(h, wv2, bv2, nullptr, vb, 8192, stream);
    attn_kernel<<<dim3(512), dim3(256), 0, stream>>>(qb, kb, vb, aO, attn2);
    launch_gemm<128, 128, true, true,  false>(aO, wo2, bo2, h, h, 8192, stream);

    // FFN with residual
    launch_gemm<128, 128, true, false, true >(h, wc1, bc1, nullptr, vb, 8192, stream);
    launch_gemm<128, 128, true, true,  false>(vb, wc2, bc2, h, h, 8192, stream);

    // node features
    nf_kernel<<<dim3(256), dim3(256), 0, stream>>>(h, wf, bf, nf);

    // edge MLP factorization: P = h @ we1[:128], Q = h @ we1[128:]
    launch_gemm<128, 128, false, false, false>(h, we1,             nullptr, nullptr, qb, 8192, stream);
    launch_gemm<128, 128, false, false, false>(h, we1 + 128 * 128, nullptr, nullptr, kb, 8192, stream);
    edge_kernel<<<dim3(128), dim3(256), 0, stream>>>(qb, kb, be1, we2, be2, edge);
}
